// SpikePopEncodingNetwork_16423954939966
// MI455X (gfx1250) — matrix operand order repaired and checked by reading, not compiled, not run on hardware
//
#include <hip/hip_runtime.h>

// ---------------- Types matching CDNA5 WMMA operand shapes ----------------
typedef _Float16 v16h __attribute__((ext_vector_type(16)));
typedef _Float16 v8h  __attribute__((ext_vector_type(8)));
typedef float    v8f  __attribute__((ext_vector_type(8)));

#define T_WIN   8
#define BN      4096      // B*N rows
#define C_DIM   64        // K dimension
#define E_DIM   3136      // output features (= 196 * 16)
#define COL_TILES 196
#define ROW_TILES 256
#define WAVES_PER_BLOCK 4

// ---------------- RNG helpers (counter-based, deterministic) ----------------
__device__ __forceinline__ unsigned int hash_u32(unsigned int x) {
    x ^= x >> 16; x *= 0x7feb352dU;
    x ^= x >> 15; x *= 0x846ca68bU;
    x ^= x >> 16;
    return x;
}

// ---------------- Kernel 1: Gaussian population encoding + Poisson ----------
// samples[t, r, c] : f16, lam = 0.5*exp(-(tau-mu)^2 / (2*0.05^2)), Knuth Poisson.
__global__ __launch_bounds__(256)
void encode_poisson_kernel(const float* __restrict__ taus,
                           _Float16* __restrict__ samples) {
    const int idx = blockIdx.x * blockDim.x + threadIdx.x;
    const int total = T_WIN * BN * C_DIM;
    if (idx >= total) return;

    const int c = idx & (C_DIM - 1);
    const int r = (idx >> 6) & (BN - 1);

    const float tau = taus[r];
    const float mu  = (float)c * (1.0f / (float)C_DIM);
    const float d   = tau - mu;
    // 1/(2*sigma^2) = 1/(2*0.0025) = 200
    const float lam = 0.5f * __expf(-d * d * 200.0f);

    // Knuth's algorithm: lam <= 0.5 so exp(-lam) >= 0.60 — short loop.
    const float L = __expf(-lam);
    unsigned int s = hash_u32((unsigned int)idx * 0x9E3779B9u + 0xC0FFEEu);
    float p = 1.0f;
    int k = 0;
#pragma unroll 1
    do {
        ++k;
        s = hash_u32(s + 0x9E3779B9u);
        p *= (float)(s >> 8) * (1.0f / 16777216.0f);
    } while (p > L && k < 16);

    samples[idx] = (_Float16)(float)(k - 1);
}

// ---------------- Kernel 1b: convert W (f32 -> f16) once --------------------
__global__ __launch_bounds__(256)
void convert_w_kernel(const float* __restrict__ W,
                      _Float16* __restrict__ Wh) {
    const int idx = blockIdx.x * blockDim.x + threadIdx.x;
    if (idx < E_DIM * C_DIM) Wh[idx] = (_Float16)W[idx];
}

// ---------------- Kernel 2: fused WMMA GEMM + LIF dynamics ------------------
// One wave owns a 16x16 tile of (rows x E). Membrane state lives in 8 f32
// VGPRs (the WMMA C/D layout) across all 8 time steps. B fragments (W^T in
// f16) and bias are loaded once and reused every step. t-loop kept rolled:
// the LIF recurrence serializes it and ~50K waves hide the latency.
__global__ __launch_bounds__(WAVES_PER_BLOCK * 32)
void lif_wmma_kernel(const _Float16* __restrict__ X,   // [T, BN, 64] f16
                     const _Float16* __restrict__ Wh,  // [E, 64] f16
                     const float* __restrict__ bias,   // [E] f32
                     float* __restrict__ out) {        // [T, BN, E] f32
    const int lane = threadIdx.x & 31;
    const int wave = threadIdx.x >> 5;

    // 49 groups of 4 col-tiles per row-tile: grid = ROW_TILES * 49
    const int colGroup = blockIdx.x % (COL_TILES / WAVES_PER_BLOCK);
    const int rowTile  = blockIdx.x / (COL_TILES / WAVES_PER_BLOCK);
    const int colTile  = colGroup * WAVES_PER_BLOCK + wave;

    const int row0 = rowTile * 16;
    const int col0 = colTile * 16;

    const int n    = lane & 15;   // column within tile (B/C/D: N = lane%16)
    const int half = lane >> 4;   // lane group 0/1

    // ---- B fragments: W^T 64x16 as two 32x16 f16 fragments, loaded once.
    // B k order used here: lanes 0-15 hold K=0..15, lanes 16-31 hold K=16..31,
    // column N = lane&15, 2 K-values per VGPR -> 16 contiguous halves/lane.
    const _Float16* wr = Wh + (size_t)(col0 + n) * C_DIM + half * 16;
    const v16h b0 = *(const v16h*)(wr);        // K block 0..31
    const v16h b1 = *(const v16h*)(wr + 32);   // K block 32..63
    const float bv = bias[col0 + n];

    // ---- Membrane state in the C/D accumulator layout.
    v8f mem;
#pragma unroll
    for (int i = 0; i < 8; ++i) mem[i] = 0.0f;

    // A-fragment addressing (16-bit A 16x32): lane m=lane&15 is row M;
    // lanes 0-15 hold K 0..15, lanes 16-31 hold K 16..31 (matches B).
    const int m  = lane & 15;
    const int kh = half * 16;

    const _Float16* xr = X + ((size_t)row0 + m) * C_DIM;      // + t*BN*C per step
    float* op = out + ((size_t)row0 + half * 8) * E_DIM + col0 + n;

#pragma unroll 1
    for (int t = 0; t < T_WIN; ++t) {
        v8h p0 = *(const v8h*)(xr + kh);        // K kh..kh+7
        v8h p1 = *(const v8h*)(xr + 8 + kh);    // K 8+kh..
        v8h p2 = *(const v8h*)(xr + 32 + kh);   // K 32+kh..
        v8h p3 = *(const v8h*)(xr + 40 + kh);   // K 40+kh..
        v16h a0 = __builtin_shufflevector(p0, p1, 0,1,2,3,4,5,6,7,8,9,10,11,12,13,14,15);
        v16h a1 = __builtin_shufflevector(p2, p3, 0,1,2,3,4,5,6,7,8,9,10,11,12,13,14,15);

        v8f acc = {};
        acc = __builtin_amdgcn_wmma_f32_16x16x32_f16(false, a0, false, b0,
                                                     (short)0, acc, false, false);
        acc = __builtin_amdgcn_wmma_f32_16x16x32_f16(false, a1, false, b1,
                                                     (short)0, acc, false, false);

        // LIF: mem += (pre - mem)/2 ; spike ; hard reset ; store spike.
        // Rows i..i+7 are op[i*E_DIM] -> immediate 24-bit offsets (i*12544 B).
#pragma unroll
        for (int i = 0; i < 8; ++i) {
            const float pre = acc[i] + bv;
            const float mv  = mem[i] + (pre - mem[i]) * 0.5f;
            const float s   = (mv > 0.5f) ? 1.0f : 0.0f;
            op[(size_t)i * E_DIM] = s;
            mem[i] = mv * (1.0f - s);
        }

        xr += (size_t)BN * C_DIM;
        op += (size_t)BN * E_DIM;
    }
}

// ---------------- Launch -----------------------------------------------------
extern "C" void kernel_launch(void* const* d_in, const int* in_sizes, int n_in,
                              void* d_out, int out_size, void* d_ws, size_t ws_size,
                              hipStream_t stream) {
    const float* taus = (const float*)d_in[0];   // [64,64] = 4096
    const float* W    = (const float*)d_in[1];   // [3136,64]
    const float* bias = (const float*)d_in[2];   // [3136]
    // d_in[3] = time_window (fixed at 8 per setup_inputs)

    _Float16* samples = (_Float16*)d_ws;                         // 4 MB
    _Float16* Wh = (_Float16*)((char*)d_ws
                   + (size_t)T_WIN * BN * C_DIM * sizeof(_Float16)); // 392 KB
    float* out = (float*)d_out;                  // [8,4096,3136] f32

    {
        const int total = T_WIN * BN * C_DIM;
        encode_poisson_kernel<<<(total + 255) / 256, 256, 0, stream>>>(taus, samples);
    }
    {
        const int total = E_DIM * C_DIM;
        convert_w_kernel<<<(total + 255) / 256, 256, 0, stream>>>(W, Wh);
    }
    {
        const int blocks = ROW_TILES * (COL_TILES / WAVES_PER_BLOCK); // 256*49
        lif_wmma_kernel<<<blocks, WAVES_PER_BLOCK * 32, 0, stream>>>(
            samples, Wh, bias, out);
    }
}
